// SkipGRU_70918499992242
// MI455X (gfx1250) — compile-verified, matrix operand order checked by reading, not run on hardware
//
#include <hip/hip_runtime.h>
#include <hip/hip_bf16.h>
#include <math.h>

// ---------------- problem constants ----------------
constexpr int B_   = 64;
constexpr int S_   = 512;
constexpr int IN_  = 512;
constexpr int H_   = 1024;
constexpr int LAT_ = 128;
constexpr int G3_  = 3 * H_;          // 3072
constexpr int NCOL = 5 * H_;          // 5120 fused output columns: r,u,gh_n,gi_n,h_hat
constexpr int KTOT = H_ + IN_;        // 1536 fused K (h | x_t)

// persistent kernel geometry: 64 n-groups x 4 M-tiles = 256 waves, all active
constexpr int PBLOCKS  = 32;
constexpr int PTHREADS = 256;

// ---------------- workspace layout (bytes) ----------------
constexpr size_t OFF_X16    = 0;                                  // f16 inputs [B,S,IN]
constexpr size_t SZ_X16     = (size_t)B_ * S_ * IN_ * 2;
constexpr size_t OFF_H16    = OFF_X16 + SZ_X16;                   // f16 h state, DOUBLE buffered
constexpr size_t SZ_H16     = (size_t)2 * B_ * H_ * 2;
constexpr size_t OFF_WHH16  = OFF_H16 + SZ_H16;                   // f16 W_hh [3H,H]
constexpr size_t SZ_WHH16   = (size_t)G3_ * H_ * 2;
constexpr size_t OFF_WLHT16 = OFF_WHH16 + SZ_WHH16;               // f16 W_lh^T [H(k),H(j)]
constexpr size_t SZ_WLHT16  = (size_t)H_ * H_ * 2;
constexpr size_t OFF_WCOMB  = OFF_WLHT16 + SZ_WLHT16;             // f16 W_comb [3H,H]
constexpr size_t SZ_WCOMB   = (size_t)G3_ * H_ * 2;
constexpr size_t OFF_WCAT   = OFF_WCOMB + SZ_WCOMB;               // f16 Wcat [NCOL,KTOT]
constexpr size_t SZ_WCAT    = (size_t)NCOL * KTOT * 2;
constexpr size_t OFF_CLIN   = OFF_WCAT + SZ_WCAT;                 // f32 c_lin [B,H]
constexpr size_t SZ_CLIN    = (size_t)B_ * H_ * 4;
constexpr size_t OFF_CB     = OFF_CLIN + SZ_CLIN;                 // f32 Cb [B,NCOL]
constexpr size_t SZ_CB      = (size_t)B_ * NCOL * 4;
constexpr size_t OFF_BAR    = OFF_CB + SZ_CB;                     // 2 x u32 barrier

typedef __attribute__((ext_vector_type(16))) _Float16 v16h;
typedef __attribute__((ext_vector_type(8)))  float    v8f;

// ---------------- small helper kernels ----------------
__global__ void k_f32_to_f16(const float* __restrict__ src, _Float16* __restrict__ dst, int n) {
    int i = blockIdx.x * blockDim.x + threadIdx.x;
    if (i < n) dst[i] = (_Float16)src[i];
}

// W_lhT16[k*H + j] = W_lin[j*(H+LAT) + k]   (transposed h-part of W_lin)
__global__ void k_build_wlhT(const float* __restrict__ W_lin, _Float16* __restrict__ WlhT) {
    int i = blockIdx.x * blockDim.x + threadIdx.x;
    if (i >= H_ * H_) return;
    int k = i >> 10, j = i & (H_ - 1);
    WlhT[i] = (_Float16)W_lin[(size_t)j * (H_ + LAT_) + k];
}

// h buffer 0 init from h0; thread 0 also zeroes the grid barrier
__global__ void k_init_h(const float* __restrict__ h0, _Float16* __restrict__ h16,
                         unsigned* __restrict__ bar) {
    int i = blockIdx.x * blockDim.x + threadIdx.x;
    if (i == 0) { bar[0] = 0u; bar[1] = 0u; }
    if (i < B_ * H_) h16[i] = (_Float16)h0[i];
}

// c_lin[b,n] = b_lin[n] + sum_l z[b,l] * W_lin[n, H + l]
__global__ void k_clin(const float* __restrict__ z, const float* __restrict__ W_lin,
                       const float* __restrict__ b_lin, float* __restrict__ clin) {
    int i = blockIdx.x * blockDim.x + threadIdx.x;
    if (i >= B_ * H_) return;
    int b = i >> 10, n = i & (H_ - 1);
    float s = b_lin[n];
    const float* wr = W_lin + (size_t)n * (H_ + LAT_) + H_;
    const float* zr = z + (size_t)b * LAT_;
    for (int l = 0; l < LAT_; ++l) s += zr[l] * wr[l];
    clin[i] = s;
}

// generic WMMA GEMM: C[M,N] (f16) = A[M,K] (f16, row-major) x Bt[N,K] (f16, N-major)
__global__ void k_wmma_gemm_f16(const _Float16* __restrict__ A, const _Float16* __restrict__ Bt,
                                _Float16* __restrict__ C, int M, int N, int K) {
    const int lane  = threadIdx.x & 31;
    const int wave  = blockIdx.x * (blockDim.x >> 5) + (threadIdx.x >> 5);
    const int nwav  = gridDim.x * (blockDim.x >> 5);
    const int row16 = lane & 15;
    const int khalf = (lane >> 4) << 4;
    const int tilesM = M >> 4, tilesN = N >> 4;
    for (int tile = wave; tile < tilesM * tilesN; tile += nwav) {
        const int tm = tile % tilesM, tn = tile / tilesM;
        const int m0 = tm << 4, n0 = tn << 4;
        v8f acc = {};
        const _Float16* arow = A  + (size_t)(m0 + row16) * K + khalf;
        const _Float16* brow = Bt + (size_t)(n0 + row16) * K + khalf;
        for (int k0 = 0; k0 < K; k0 += 32) {
            v16h a = *(const v16h*)(arow + k0);
            v16h b = *(const v16h*)(brow + k0);
            acc = __builtin_amdgcn_wmma_f32_16x16x32_f16(false, a, false, b,
                                                         (short)0, acc, false, false);
        }
        const int cm = m0 + ((lane >> 4) << 3);
        const int cn = n0 + row16;
        #pragma unroll
        for (int v = 0; v < 8; ++v)
            C[(size_t)(cm + v) * N + cn] = (_Float16)acc[v];
    }
}

// Build fused weight Wcat[n*KTOT + k] (N-major, K-contiguous)
__global__ void k_build_wcat(const _Float16* __restrict__ Wcomb, const float* __restrict__ W_ih,
                             const float* __restrict__ W_lin, _Float16* __restrict__ Wcat) {
    int i = blockIdx.x * blockDim.x + threadIdx.x;
    if (i >= NCOL * KTOT) return;
    int n = i / KTOT, k = i - n * KTOT;
    int sec = n >> 10, np = n & (H_ - 1);
    _Float16 v = (_Float16)0.0f;
    if (sec <= 1) {                       // r / u gates: gh + gi fused
        v = (k < H_) ? Wcomb[(size_t)(sec * H_ + np) * H_ + k]
                     : (_Float16)W_ih[(size_t)(sec * H_ + np) * IN_ + (k - H_)];
    } else if (sec == 2) {                // gh_n (hidden side of n-gate)
        if (k < H_) v = Wcomb[(size_t)(2 * H_ + np) * H_ + k];
    } else if (sec == 3) {                // gi_n (input side of n-gate)
        if (k >= H_) v = (_Float16)W_ih[(size_t)(2 * H_ + np) * IN_ + (k - H_)];
    } else {                              // h_hat = h @ W_lh^T
        if (k < H_) v = (_Float16)W_lin[(size_t)np * (H_ + LAT_) + k];
    }
    Wcat[i] = v;
}

// Bias/constant matrix Cb[b, NCOL]
__global__ void k_build_cb(const float* __restrict__ clin, const float* __restrict__ W_hh,
                           const float* __restrict__ b_ih, const float* __restrict__ b_hh,
                           float* __restrict__ Cb) {
    int i = blockIdx.x * blockDim.x + threadIdx.x;
    if (i >= B_ * NCOL) return;
    int b = i / NCOL, n = i - b * NCOL;
    int sec = n >> 10, np = n & (H_ - 1);
    float s;
    if (sec <= 2) {                       // c_lin @ W_hh_sec^T + biases
        const float* cr = clin + (size_t)b * H_;
        const float* wr = W_hh + (size_t)(sec * H_ + np) * H_;
        s = 0.0f;
        for (int j = 0; j < H_; ++j) s += cr[j] * wr[j];
        s += b_hh[sec * H_ + np];
        if (sec <= 1) s += b_ih[sec * H_ + np];
    } else if (sec == 3) {
        s = b_ih[2 * H_ + np];
    } else {
        s = clin[(size_t)b * H_ + np];
    }
    Cb[i] = s;
}

// ---------------- device-wide barrier ----------------
__device__ __forceinline__ void grid_barrier(volatile unsigned* bar, unsigned nblk) {
    __threadfence();
    __syncthreads();
    if (threadIdx.x == 0) {
        unsigned gen = bar[1];
        unsigned prev = atomicAdd((unsigned*)&bar[0], 1u);
        if (prev == nblk - 1u) {
            bar[0] = 0u;
            __threadfence();
            atomicAdd((unsigned*)&bar[1], 1u);
        } else {
            while (bar[1] == gen) { __builtin_amdgcn_s_sleep(1); }
        }
    }
    __syncthreads();
}

__device__ __forceinline__ float sigmoidf_(float x) { return 1.0f / (1.0f + __expf(-x)); }

#define WMMA_F16(a, b, c) \
    __builtin_amdgcn_wmma_f32_16x16x32_f16(false, (a), false, (b), (short)0, (c), false, false)

// ---------------- persistent recurrent kernel ----------------
// Each wave owns one 16-row M-tile x one 16-col n-group, and accumulates the
// SAME tile position in all 5 output sections (r, u, gh_n, gi_n, h_hat) in 5
// register accumulators. The GRU gate math is then purely element-wise across
// the 5 accs (identical C/D lane layout), so the epilogue runs in-register:
// no D buffer, no second grid barrier. h is double-buffered across steps.
__global__ void __launch_bounds__(PTHREADS)
k_skipgru(const _Float16* __restrict__ x16, _Float16* __restrict__ h16,
          const _Float16* __restrict__ Wcat, const float* __restrict__ Cb,
          float* __restrict__ out, unsigned* __restrict__ bar) {
    const int lane  = threadIdx.x & 31;
    const int wv    = (blockIdx.x << 3) + (threadIdx.x >> 5);   // 0..255
    const int row16 = lane & 15;
    const int khalf = (lane >> 4) << 4;
    const int m0    = (wv & 3) << 4;                            // M-tile (batch rows)
    const int n0    = (wv >> 2) << 4;                           // n-group within H

    // Wcat row pointers for the 5 sections (constant across steps)
    const _Float16* brow0 = Wcat + (size_t)(0 * H_ + n0 + row16) * KTOT + khalf;
    const _Float16* brow1 = Wcat + (size_t)(1 * H_ + n0 + row16) * KTOT + khalf;
    const _Float16* brow2 = Wcat + (size_t)(2 * H_ + n0 + row16) * KTOT + khalf;
    const _Float16* brow3 = Wcat + (size_t)(3 * H_ + n0 + row16) * KTOT + khalf;
    const _Float16* brow4 = Wcat + (size_t)(4 * H_ + n0 + row16) * KTOT + khalf;

    const int arow = m0 + row16;                                // batch row this lane loads
    const size_t HB = (size_t)B_ * H_;

    for (int t = 0; t < S_; ++t) {
        const _Float16* hcur  = h16 + ((t & 1) ? HB : 0);
        _Float16*       hnext = h16 + ((t & 1) ? 0 : HB);
        const _Float16* ha = hcur + (size_t)arow * H_ + khalf;
        const _Float16* xa = x16 + ((size_t)arow * S_ + t) * IN_ + khalf;

        v8f acc0 = {}, acc1 = {}, acc2 = {}, acc3 = {}, acc4 = {};
        for (int k0 = 0; k0 < H_; k0 += 32) {                   // hidden K-slab
            v16h a  = *(const v16h*)(ha + k0);
            v16h b0 = *(const v16h*)(brow0 + k0);
            v16h b1 = *(const v16h*)(brow1 + k0);
            v16h b2 = *(const v16h*)(brow2 + k0);
            v16h b3 = *(const v16h*)(brow3 + k0);
            v16h b4 = *(const v16h*)(brow4 + k0);
            acc0 = WMMA_F16(a, b0, acc0);
            acc1 = WMMA_F16(a, b1, acc1);
            acc2 = WMMA_F16(a, b2, acc2);
            acc3 = WMMA_F16(a, b3, acc3);
            acc4 = WMMA_F16(a, b4, acc4);
        }
        for (int k0 = 0; k0 < IN_; k0 += 32) {                  // input K-slab
            v16h a  = *(const v16h*)(xa + k0);
            v16h b0 = *(const v16h*)(brow0 + H_ + k0);
            v16h b1 = *(const v16h*)(brow1 + H_ + k0);
            v16h b2 = *(const v16h*)(brow2 + H_ + k0);
            v16h b3 = *(const v16h*)(brow3 + H_ + k0);
            v16h b4 = *(const v16h*)(brow4 + H_ + k0);
            acc0 = WMMA_F16(a, b0, acc0);
            acc1 = WMMA_F16(a, b1, acc1);
            acc2 = WMMA_F16(a, b2, acc2);
            acc3 = WMMA_F16(a, b3, acc3);
            acc4 = WMMA_F16(a, b4, acc4);
            if (t + 1 < S_)                                     // stream next step's x
                __builtin_prefetch(xa + IN_ + k0, 0, 3);
        }

        // -------- in-register gate epilogue --------
        const int cmo = (lane >> 4) << 3;                       // 0 or 8
        const int n   = n0 + row16;                             // column within H
        #pragma unroll
        for (int v = 0; v < 8; ++v) {
            const int b = m0 + cmo + v;
            const float* Cb_b = Cb + (size_t)b * NCOL + n;
            const float r    = sigmoidf_(acc0[v] + Cb_b[0 * H_]);
            const float u    = sigmoidf_(acc1[v] + Cb_b[1 * H_]);
            const float hn   = acc2[v] + Cb_b[2 * H_];
            const float in_  = acc3[v] + Cb_b[3 * H_];
            const float hhat = acc4[v] + Cb_b[4 * H_];
            const float nv   = tanhf(in_ + r * hn);
            const float hnew = (1.0f - u) * nv + u * hhat;
            out[((size_t)b * S_ + t) * H_ + n] = hnew;          // outs[B,S,H]
            hnext[(size_t)b * H_ + n] = (_Float16)hnew;
            if (t == S_ - 1)
                out[(size_t)B_ * S_ * H_ + (size_t)b * H_ + n] = hnew;  // hT
        }
        grid_barrier(bar, (unsigned)gridDim.x);                 // ONE barrier per step
    }
}

// ---------------- host launcher ----------------
extern "C" void kernel_launch(void* const* d_in, const int* in_sizes, int n_in,
                              void* d_out, int out_size, void* d_ws, size_t ws_size,
                              hipStream_t stream) {
    (void)in_sizes; (void)n_in; (void)out_size; (void)ws_size;
    const float* inputs = (const float*)d_in[0];
    const float* h0     = (const float*)d_in[1];
    const float* z      = (const float*)d_in[2];
    const float* W_lin  = (const float*)d_in[3];
    const float* b_lin  = (const float*)d_in[4];
    const float* W_ih   = (const float*)d_in[5];
    const float* b_ih   = (const float*)d_in[6];
    const float* W_hh   = (const float*)d_in[7];
    const float* b_hh   = (const float*)d_in[8];

    char* ws = (char*)d_ws;
    _Float16* x16    = (_Float16*)(ws + OFF_X16);
    _Float16* h16    = (_Float16*)(ws + OFF_H16);
    _Float16* Whh16  = (_Float16*)(ws + OFF_WHH16);
    _Float16* WlhT16 = (_Float16*)(ws + OFF_WLHT16);
    _Float16* Wcomb  = (_Float16*)(ws + OFF_WCOMB);
    _Float16* Wcat   = (_Float16*)(ws + OFF_WCAT);
    float*    clin   = (float*)(ws + OFF_CLIN);
    float*    Cb     = (float*)(ws + OFF_CB);
    unsigned* bar    = (unsigned*)(ws + OFF_BAR);
    float*    out    = (float*)d_out;

    const int TB = 256;
    // one-time precompute (all inside the captured graph, deterministic)
    k_f32_to_f16<<<(B_ * S_ * IN_ + TB - 1) / TB, TB, 0, stream>>>(inputs, x16, B_ * S_ * IN_);
    k_f32_to_f16<<<(G3_ * H_ + TB - 1) / TB, TB, 0, stream>>>(W_hh, Whh16, G3_ * H_);
    k_build_wlhT<<<(H_ * H_ + TB - 1) / TB, TB, 0, stream>>>(W_lin, WlhT16);
    k_init_h<<<(B_ * H_ + TB - 1) / TB, TB, 0, stream>>>(h0, h16, bar);
    k_clin<<<(B_ * H_ + TB - 1) / TB, TB, 0, stream>>>(z, W_lin, b_lin, clin);
    // W_comb[3H,H] = W_hh[3H,H] x W_lh[H,H]  (f16 WMMA, f32 accumulate)
    k_wmma_gemm_f16<<<256, TB, 0, stream>>>(Whh16, WlhT16, Wcomb, G3_, H_, H_);
    k_build_wcat<<<(NCOL * KTOT + TB - 1) / TB, TB, 0, stream>>>(Wcomb, W_ih, W_lin, Wcat);
    k_build_cb<<<(B_ * NCOL + TB - 1) / TB, TB, 0, stream>>>(clin, W_hh, b_ih, b_hh, Cb);
    // persistent recurrent kernel: 512 steps, fused GEMM+gates, 1 barrier/step
    k_skipgru<<<PBLOCKS, PTHREADS, 0, stream>>>(x16, h16, Wcat, Cb, out, bar);
}
